// RHMemoryDecoder_85246510891662
// MI455X (gfx1250) — matrix operand order repaired and checked by reading, not compiled, not run on hardware
//
#include <hip/hip_runtime.h>
#include <hip/hip_bf16.h>
#include <math.h>

#define B_  16
#define S_  512
#define D_  256
#define H_  8
#define HD_ 32
#define FF_ 1024
#define L_  2
#define NS_ 1024
#define BS_ (B_*S_)    /* 8192  */
#define SD_ (S_*D_)    /* 131072 */
#define KSPLIT_ 64

typedef __attribute__((ext_vector_type(16))) _Float16 v16h;
typedef __attribute__((ext_vector_type(8)))  _Float16 v8h;
typedef __attribute__((ext_vector_type(8)))  float    v8f;
typedef __attribute__((ext_vector_type(4)))  unsigned int v4u;
typedef __attribute__((ext_vector_type(8)))  unsigned int v8u;

// Load a 16-half WMMA fragment as two contiguous 8-half (16B) vectors.
// A-matrix 16x32 f16 layout: lane row=lane&15, kh=lane>>4;
//   elems 0..7  -> K = k0 + kh*8 + e        (contiguous)
//   elems 8..15 -> K = k0 + 16 + kh*8 + e-8 (contiguous)
// B-matrix 32x16 f16 layout: lane col=lane&15;
//   elems 0..15 -> same K order as A (kh*8 + e, then 16 + kh*8 + e-8)
__device__ __forceinline__ v16h load_frag32(const _Float16* p0, const _Float16* p1) {
  v8h lo = *(const v8h*)p0;
  v8h hi = *(const v8h*)p1;
  v16h r;
#pragma unroll
  for (int i = 0; i < 8; ++i) { r[i] = lo[i]; r[8 + i] = hi[i]; }
  return r;
}

enum { EPI_F16 = 0, EPI_F32 = 1, EPI_GELU_F16 = 2, EPI_RESID = 3, EPI_RESID_MASK = 4 };

// C = A[M,K](f16,row-major) x W (stored transposed [N,K] f16) + bias.
// Block = 4 waves, block tile 32(M) x 256(N); wave tile 32x64 (8 WMMAs/k-step,
// B fragment reused across both M halves). The 32xK A panel is DMA'd into LDS
// once per block by the Tensor Data Mover (issued from wave 0 only via a
// scalar branch, since TDM ignores EXEC), completion via s_wait_tensorcnt.
template<int EPI, int K>
__global__ __launch_bounds__(128)
void gemm_wmma(const _Float16* __restrict__ A, const _Float16* __restrict__ Wt,
               const float* __restrict__ bias,
               float* __restrict__ outF, _Float16* __restrict__ outH,
               const unsigned char* __restrict__ mask, int N)
{
  __shared__ _Float16 Atile[32 * K];   // sole LDS object -> offset 0
  const int lane = threadIdx.x & 31;
  const int m0 = blockIdx.x * 32;
  const int row = lane & 15;
  const int kh  = lane >> 4;

  // --- TDM: async-load A[m0..m0+31][0..K) into LDS (2D descriptor) ---
  const int wid = __builtin_amdgcn_readfirstlane((int)threadIdx.x) >> 5;
  if (wid == 0) {
    const unsigned long long ga =
        (unsigned long long)(const void*)(A + (size_t)m0 * K);
    v4u g0;
    g0[0] = 1u;                                   // count=1 (valid D#)
    g0[1] = 0u;                                   // lds_addr = 0 (Atile base)
    g0[2] = (unsigned)ga;                         // global_addr[31:0]
    g0[3] = ((unsigned)(ga >> 32) & 0x01FFFFFFu)  // global_addr[56:32]
          | (2u << 30);                           // type = 2 ("image")
    v8u g1;
    g1[0] = 1u << 16;                             // data_size=1 (2 bytes)
    g1[1] = ((unsigned)K & 0xFFFFu) << 16;        // tensor_dim0[15:0] = K
    g1[2] = 32u << 16;                            // tensor_dim0 hi=0; tensor_dim1 lo=32
    g1[3] = ((unsigned)K) << 16;                  // tensor_dim1 hi=0; tile_dim0 = K
    g1[4] = 32u;                                  // tile_dim1 = 32; tile_dim2 = 0
    g1[5] = (unsigned)K;                          // tensor_dim0_stride[31:0] = K
    g1[6] = 0u;                                   // stride hi, dim1_stride lo
    g1[7] = 0u;
    v4u g2 = {0u, 0u, 0u, 0u};
    v4u g3 = {0u, 0u, 0u, 0u};
    asm volatile("tensor_load_to_lds %0, %1, %2, %3"
                 :: "s"(g0), "s"(g1), "s"(g2), "s"(g3)
                 : "memory");
    __builtin_amdgcn_s_wait_tensorcnt(0);
  }
  __syncthreads();

  // --- compute: wave w -> columns [w*64, w*64+64) ---
  const int n0 = blockIdx.y * 256 + (threadIdx.x >> 5) * 64;
  v8f acc[2][4] = {};
  const _Float16* a0p = &Atile[(size_t)row * K];
  const _Float16* a1p = &Atile[(size_t)(16 + row) * K];
  for (int k0 = 0; k0 < K; k0 += 32) {
    v16h a0 = load_frag32(a0p + k0 + kh * 8, a0p + k0 + 16 + kh * 8);
    v16h a1 = load_frag32(a1p + k0 + kh * 8, a1p + k0 + 16 + kh * 8);
#pragma unroll
    for (int t = 0; t < 4; ++t) {
      const _Float16* bp = Wt + (size_t)(n0 + t * 16 + row) * K + k0 + kh * 8;
      v16h b = load_frag32(bp, bp + 16);
      acc[0][t] = __builtin_amdgcn_wmma_f32_16x16x32_f16(false, a0, false, b,
                                                         (short)0, acc[0][t], false, false);
      acc[1][t] = __builtin_amdgcn_wmma_f32_16x16x32_f16(false, a1, false, b,
                                                         (short)0, acc[1][t], false, false);
    }
  }
  // C/D layout: elem r -> M = m0 + half*16 + r + 8*kh, N = n0 + t*16 + (lane&15)
#pragma unroll
  for (int half = 0; half < 2; ++half) {
#pragma unroll
    for (int t = 0; t < 4; ++t) {
      const int n = n0 + t * 16 + row;
      const float bv = bias[n];
#pragma unroll
      for (int r = 0; r < 8; ++r) {
        const int m = m0 + half * 16 + r + (kh << 3);
        float v = acc[half][t][r] + bv;
        const size_t offn = (size_t)m * N + n;
        if (EPI == EPI_F16) {
          outH[offn] = (_Float16)v;
        } else if (EPI == EPI_F32) {
          outF[offn] = v;
        } else if (EPI == EPI_GELU_F16) {
          v = 0.5f * v * (1.0f + erff(v * 0.70710678118654752f));
          outH[offn] = (_Float16)v;
        } else if (EPI == EPI_RESID) {
          outF[offn] += v;
        } else if (EPI == EPI_RESID_MASK) {
          outF[offn] += mask[m] ? v : 0.0f;
        }
      }
    }
  }
}

// One wave per (b, h, 16-query tile). Q.K^T (single WMMA k-step, K=HD=32),
// fp32 softmax via LDS, then P.V with V pre-transposed [B,H,HD,S].
__global__ __launch_bounds__(32)
void attn_wmma(const _Float16* __restrict__ q16, const _Float16* __restrict__ k16,
               const _Float16* __restrict__ vt16, const unsigned char* __restrict__ mask,
               _Float16* __restrict__ ctx16)
{
  __shared__ float    sc[16 * S_];   // 32 KB
  __shared__ _Float16 pw[16 * S_];   // 16 KB
  const int lane = threadIdx.x & 31;
  const int row = lane & 15, kh = lane >> 4;
  const int q0 = blockIdx.x * 16;
  const int h  = blockIdx.y;
  const int b  = blockIdx.z;
  const size_t bh = (size_t)(b * H_ + h);
  const float scale = 0.17677669529663687f; // 1/sqrt(HD)

  const _Float16* qbase = q16 + (bh * S_ + q0) * HD_;
  const _Float16* kbase = k16 + bh * S_ * HD_;
  v16h aq = load_frag32(qbase + row * HD_ + kh * 8, qbase + row * HD_ + 16 + kh * 8);

  for (int kt = 0; kt < S_ / 16; ++kt) {
    const int key = kt * 16 + row;
    const _Float16* bp = kbase + (size_t)key * HD_ + kh * 8;
    v16h bf = load_frag32(bp, bp + 16);
    v8f s = {};
    s = __builtin_amdgcn_wmma_f32_16x16x32_f16(false, aq, false, bf, (short)0, s, false, false);
    const float mfac = mask[b * S_ + key] ? 0.0f : -3.0e38f;
#pragma unroll
    for (int r = 0; r < 8; ++r) {
      const int qq = r + (kh << 3);
      sc[qq * S_ + key] = s[r] * scale + mfac;
    }
  }
  __syncthreads();
  if (lane < 16) {
    float* rowp = sc + lane * S_;
    float mx = -3.0e38f;
    for (int j = 0; j < S_; ++j) mx = fmaxf(mx, rowp[j]);
    float sum = 0.0f;
    for (int j = 0; j < S_; ++j) { float e = __expf(rowp[j] - mx); sum += e; rowp[j] = e; }
    const float inv = 1.0f / sum;
    _Float16* pp = pw + lane * S_;
    for (int j = 0; j < S_; ++j) pp[j] = (_Float16)(rowp[j] * inv);
  }
  __syncthreads();

  v8f c[2] = {};
  const _Float16* vtb = vt16 + bh * HD_ * S_;
  for (int kt = 0; kt < S_ / 32; ++kt) {
    const int kb2 = kt * 32;
    const _Float16* pr = pw + row * S_ + kb2 + kh * 8;
    v16h a = load_frag32(pr, pr + 16);
#pragma unroll
    for (int t = 0; t < 2; ++t) {
      const _Float16* bp = vtb + (size_t)(t * 16 + row) * S_ + kb2 + kh * 8;
      v16h bf = load_frag32(bp, bp + 16);
      c[t] = __builtin_amdgcn_wmma_f32_16x16x32_f16(false, a, false, bf, (short)0, c[t], false, false);
    }
  }
#pragma unroll
  for (int t = 0; t < 2; ++t)
#pragma unroll
    for (int r = 0; r < 8; ++r) {
      const int m = b * S_ + q0 + r + (kh << 3);
      const int d = h * HD_ + t * 16 + row;
      ctx16[(size_t)m * D_ + d] = (_Float16)c[t][r];
    }
}

template<int OUT16>
__global__ __launch_bounds__(256)
void layernorm_k(const float* __restrict__ x, const float* __restrict__ gam,
                 const float* __restrict__ bet, float* __restrict__ outF,
                 _Float16* __restrict__ outH)
{
  __shared__ float red[256];
  const int m = blockIdx.x, t = threadIdx.x;
  const float v = x[(size_t)m * D_ + t];
  red[t] = v; __syncthreads();
  for (int st = 128; st > 0; st >>= 1) { if (t < st) red[t] += red[t + st]; __syncthreads(); }
  const float mean = red[0] * (1.0f / D_);
  __syncthreads();
  const float d = v - mean;
  red[t] = d * d; __syncthreads();
  for (int st = 128; st > 0; st >>= 1) { if (t < st) red[t] += red[t + st]; __syncthreads(); }
  const float var = red[0] * (1.0f / D_);
  const float o = d * rsqrtf(var + 1e-5f) * gam[t] + bet[t];
  if (OUT16) outH[(size_t)m * D_ + t] = (_Float16)o;
  else       outF[(size_t)m * D_ + t] = o;
}

__global__ __launch_bounds__(256)
void inproj_k(const float* __restrict__ values, const float* __restrict__ dib,
              const float* __restrict__ gamma, const float* __restrict__ mtyp,
              const float* __restrict__ in_w, const float* __restrict__ in_b,
              float* __restrict__ x)
{
  const size_t idx = (size_t)blockIdx.x * 256 + threadIdx.x;
  if (idx >= (size_t)BS_ * D_) return;
  const int m = (int)(idx / D_), d = (int)(idx % D_);
  const float val = values[m];
  const float f0 = (val >= 0.0f) ? 1.0f : -1.0f;
  const float f1 = fabsf(val);
  const float f2 = gamma[m];
  const float f3 = dib[m];
  const float f4 = mtyp[m] * 2.0f - 1.0f;
  float acc = in_b[d];
  acc += f0 * in_w[0 * D_ + d] + f1 * in_w[1 * D_ + d] + f2 * in_w[2 * D_ + d]
       + f3 * in_w[3 * D_ + d] + f4 * in_w[4 * D_ + d];
  x[idx] = acc;
}

__global__ __launch_bounds__(256)
void rope_pack_k(const float* __restrict__ qf, const float* __restrict__ kf,
                 const float* __restrict__ vf, _Float16* __restrict__ q16,
                 _Float16* __restrict__ k16, _Float16* __restrict__ vt16)
{
  const size_t idx = (size_t)blockIdx.x * 256 + threadIdx.x;
  if (idx >= (size_t)BS_ * D_) return;
  const int m = (int)(idx / D_), d = (int)(idx % D_);
  const int b = m / S_, s = m % S_;
  const int h = d / HD_, hd = d % HD_;
  const float invf = __powf(10000.0f, -(float)(2 * (hd & 15)) / (float)HD_);
  const float ang = (float)s * invf;
  const float c = cosf(ang), sn = sinf(ang);
  const int pair = h * HD_ + ((hd < 16) ? hd + 16 : hd - 16);
  const size_t base = (size_t)m * D_;
  const float qv = qf[idx], qp = qf[base + pair];
  const float kv = kf[idx], kp = kf[base + pair];
  const float qr = (hd < 16) ? (qv * c - qp * sn) : (qv * c + qp * sn);
  const float kr = (hd < 16) ? (kv * c - kp * sn) : (kv * c + kp * sn);
  const size_t bh = (size_t)(b * H_ + h);
  q16[(bh * S_ + s) * HD_ + hd] = (_Float16)qr;
  k16[(bh * S_ + s) * HD_ + hd] = (_Float16)kr;
  vt16[(bh * HD_ + hd) * S_ + s] = (_Float16)vf[idx];
}

__global__ __launch_bounds__(256)
void convT_k(const float* __restrict__ W, _Float16* __restrict__ WT, int K, int N)
{
  const size_t idx = (size_t)blockIdx.x * 256 + threadIdx.x;
  if (idx >= (size_t)K * N) return;
  const int k = (int)(idx / N), n = (int)(idx % N);
  WT[(size_t)n * K + k] = (_Float16)W[idx];
}

// Bandwidth-bound head GEMM: out[b,n] = sum_k X[b,k]*W[k,n]. fp32 streaming,
// K-split for parallelism, X tile staged in LDS for lane-broadcast reuse.
__global__ __launch_bounds__(256)
void head_partial_k(const float* __restrict__ X, const float* __restrict__ W,
                    float* __restrict__ partial)
{
  __shared__ float xs[16 * 256];
  const int t = threadIdx.x;
  const int n = blockIdx.x * 256 + t;
  const int ks = blockIdx.y;
  const int KCH = SD_ / KSPLIT_; // 2048
  const int kbeg = ks * KCH;
  float acc[16];
#pragma unroll
  for (int bb = 0; bb < 16; ++bb) acc[bb] = 0.0f;
  for (int kc = 0; kc < KCH; kc += 256) {
    const int k0 = kbeg + kc;
#pragma unroll
    for (int bb = 0; bb < 16; ++bb) xs[bb * 256 + t] = X[(size_t)bb * SD_ + k0 + t];
    __syncthreads();
    for (int kk = 0; kk < 256; ++kk) {
      const float w = W[(size_t)(k0 + kk) * NS_ + n];
#pragma unroll
      for (int bb = 0; bb < 16; ++bb) acc[bb] += xs[bb * 256 + kk] * w;
    }
    __syncthreads();
  }
#pragma unroll
  for (int bb = 0; bb < 16; ++bb)
    partial[((size_t)ks * 16 + bb) * NS_ + n] = acc[bb];
}

__global__ __launch_bounds__(256)
void head_reduce_k(const float* __restrict__ partial, const float* __restrict__ hb,
                   float* __restrict__ out)
{
  const int idx = blockIdx.x * 256 + threadIdx.x;
  if (idx >= B_ * NS_) return;
  const int bb = idx / NS_, n = idx % NS_;
  float s = hb[n];
  for (int ks = 0; ks < KSPLIT_; ++ks)
    s += partial[((size_t)ks * 16 + bb) * NS_ + n];
  out[idx] = s;
}

extern "C" void kernel_launch(void* const* d_in, const int* in_sizes, int n_in,
                              void* d_out, int out_size, void* d_ws, size_t ws_size,
                              hipStream_t stream)
{
  (void)in_sizes; (void)n_in; (void)out_size; (void)ws_size;
  const float* values = (const float*)d_in[0];
  const float* dib    = (const float*)d_in[1];
  const float* gamv   = (const float*)d_in[2];
  const float* mtyp   = (const float*)d_in[3];
  const unsigned char* mask = (const unsigned char*)d_in[4];
  const float* in_w  = (const float*)d_in[5];
  const float* in_b  = (const float*)d_in[6];
  const float* ln1_s = (const float*)d_in[7];
  const float* ln1_b = (const float*)d_in[8];
  const float* qw = (const float*)d_in[9];
  const float* qb = (const float*)d_in[10];
  const float* kw = (const float*)d_in[11];
  const float* kb = (const float*)d_in[12];
  const float* vw = (const float*)d_in[13];
  const float* vb = (const float*)d_in[14];
  const float* ow = (const float*)d_in[15];
  const float* ob = (const float*)d_in[16];
  const float* ln2_s = (const float*)d_in[17];
  const float* ln2_b = (const float*)d_in[18];
  const float* ffw1 = (const float*)d_in[19];
  const float* ffb1 = (const float*)d_in[20];
  const float* ffw2 = (const float*)d_in[21];
  const float* ffb2 = (const float*)d_in[22];
  const float* fn_s = (const float*)d_in[23];
  const float* fn_b = (const float*)d_in[24];
  const float* head_w = (const float*)d_in[25];
  const float* head_b = (const float*)d_in[26];
  float* out = (float*)d_out;

  char* ws = (char*)d_ws;
  size_t off = 0;
  auto carve = [&](size_t bytes) -> char* {
    char* p = ws + off;
    off = (off + bytes + 255) & ~(size_t)255;
    return p;
  };
  float*    x     = (float*)   carve((size_t)BS_ * D_ * 4);
  _Float16* h16   = (_Float16*)carve((size_t)BS_ * D_ * 2);
  float*    qf    = (float*)   carve((size_t)BS_ * D_ * 4);
  float*    kf    = (float*)   carve((size_t)BS_ * D_ * 4);
  float*    vf    = (float*)   carve((size_t)BS_ * D_ * 4);
  _Float16* q16   = (_Float16*)carve((size_t)BS_ * D_ * 2);
  _Float16* k16   = (_Float16*)carve((size_t)BS_ * D_ * 2);
  _Float16* vt16  = (_Float16*)carve((size_t)BS_ * D_ * 2);
  _Float16* ctx16 = (_Float16*)carve((size_t)BS_ * D_ * 2);
  _Float16* ff16  = (_Float16*)carve((size_t)BS_ * FF_ * 2);
  _Float16* qwT   = (_Float16*)carve((size_t)L_ * D_ * D_ * 2);
  _Float16* kwT   = (_Float16*)carve((size_t)L_ * D_ * D_ * 2);
  _Float16* vwT   = (_Float16*)carve((size_t)L_ * D_ * D_ * 2);
  _Float16* owT   = (_Float16*)carve((size_t)L_ * D_ * D_ * 2);
  _Float16* f1T   = (_Float16*)carve((size_t)L_ * D_ * FF_ * 2);
  _Float16* f2T   = (_Float16*)carve((size_t)L_ * D_ * FF_ * 2);
  float*    part  = (float*)   carve((size_t)KSPLIT_ * B_ * NS_ * 4);
  float*    xfin  = qf; // reuse after q/k/v are consumed

  const int TPB = 256;
  const int gDD = (D_ * D_) / TPB;
  const int gDF = (D_ * FF_) / TPB;
  for (int l = 0; l < L_; ++l) {
    const size_t oDD = (size_t)l * D_ * D_;
    const size_t oDF = (size_t)l * D_ * FF_;
    convT_k<<<gDD, TPB, 0, stream>>>(qw + oDD, qwT + oDD, D_, D_);
    convT_k<<<gDD, TPB, 0, stream>>>(kw + oDD, kwT + oDD, D_, D_);
    convT_k<<<gDD, TPB, 0, stream>>>(vw + oDD, vwT + oDD, D_, D_);
    convT_k<<<gDD, TPB, 0, stream>>>(ow + oDD, owT + oDD, D_, D_);
    convT_k<<<gDF, TPB, 0, stream>>>(ffw1 + oDF, f1T + oDF, D_, FF_);
    convT_k<<<gDF, TPB, 0, stream>>>(ffw2 + oDF, f2T + oDF, FF_, D_);
  }

  const size_t nel = (size_t)BS_ * D_;
  inproj_k<<<(unsigned)((nel + TPB - 1) / TPB), TPB, 0, stream>>>(
      values, dib, gamv, mtyp, in_w, in_b, x);

  for (int l = 0; l < L_; ++l) {
    const size_t oDD = (size_t)l * D_ * D_;
    const size_t oDF = (size_t)l * D_ * FF_;
    layernorm_k<1><<<BS_, 256, 0, stream>>>(x, ln1_s + l * D_, ln1_b + l * D_, nullptr, h16);

    dim3 g(BS_ / 32, 1);
    gemm_wmma<EPI_F32, D_><<<g, 128, 0, stream>>>(h16, qwT + oDD, qb + l * D_, qf, nullptr, nullptr, D_);
    gemm_wmma<EPI_F32, D_><<<g, 128, 0, stream>>>(h16, kwT + oDD, kb + l * D_, kf, nullptr, nullptr, D_);
    gemm_wmma<EPI_F32, D_><<<g, 128, 0, stream>>>(h16, vwT + oDD, vb + l * D_, vf, nullptr, nullptr, D_);

    rope_pack_k<<<(unsigned)((nel + TPB - 1) / TPB), TPB, 0, stream>>>(qf, kf, vf, q16, k16, vt16);

    dim3 ga(S_ / 16, H_, B_);
    attn_wmma<<<ga, 32, 0, stream>>>(q16, k16, vt16, mask, ctx16);

    gemm_wmma<EPI_RESID_MASK, D_><<<g, 128, 0, stream>>>(ctx16, owT + oDD, ob + l * D_, x, nullptr, mask, D_);

    layernorm_k<1><<<BS_, 256, 0, stream>>>(x, ln2_s + l * D_, ln2_b + l * D_, nullptr, h16);

    dim3 g1(BS_ / 32, FF_ / 256);
    gemm_wmma<EPI_GELU_F16, D_><<<g1, 128, 0, stream>>>(h16, f1T + oDF, ffb1 + l * FF_, nullptr, ff16, nullptr, FF_);
    gemm_wmma<EPI_RESID, FF_><<<g, 128, 0, stream>>>(ff16, f2T + oDF, ffb2 + l * D_, x, nullptr, nullptr, D_);
  }

  layernorm_k<0><<<BS_, 256, 0, stream>>>(x, fn_s, fn_b, xfin, nullptr);

  dim3 gh(NS_ / 256, KSPLIT_);
  head_partial_k<<<gh, 256, 0, stream>>>(xfin, head_w, part);
  head_reduce_k<<<(B_ * NS_ + TPB - 1) / TPB, TPB, 0, stream>>>(part, head_b, out);
}
